// SGMHead_87771951661406
// MI455X (gfx1250) — matrix operand order repaired and checked by reading, not compiled, not run on hardware
//
#include <hip/hip_runtime.h>
#include <hip/hip_bf16.h>

// ---------------------------------------------------------------------------
// CDNA5 (gfx1250) WMMA types
// ---------------------------------------------------------------------------
typedef __attribute__((ext_vector_type(8)))  __bf16 v8bf;
typedef __attribute__((ext_vector_type(16))) __bf16 v16bf;
typedef __attribute__((ext_vector_type(8)))  float  v8f;

union BFrag {
    v16bf v;
    v8bf  h[2];
};

// ---------------------------------------------------------------------------
// Problem constants
// ---------------------------------------------------------------------------
#define BB   32
#define NN   1024
#define DVV  512
#define LL   256
#define SS   8
#define DTT  512
#define VOC  1024
#define BLROWS (BB * LL)        // 8192
#define VISROWS (BB * NN)       // 32768

// ---------------------------------------------------------------------------
// Kernel 1: embedding gather + depthwise conv1d (k=3, SAME) + SiLU
//   gsum[bl,d]  = sum_s silu(h[bl,d,s])   (feeds pointwise GEMM, mean-commuted)
//   mbase[bl,d] = sum_s E[bl,s,d]         (residual part of the mean)
// ---------------------------------------------------------------------------
__global__ __launch_bounds__(256)
void embed_dwconv_kernel(const int* __restrict__ ids,
                         const float* __restrict__ emb,
                         const float* __restrict__ dwk,
                         float* __restrict__ gsum,
                         float* __restrict__ mbase) {
    const int bl = blockIdx.x;
    __shared__ int sid[SS];
    if (threadIdx.x < SS) sid[threadIdx.x] = ids[bl * SS + threadIdx.x];
    __syncthreads();

    for (int d = threadIdx.x; d < DTT; d += 256) {
        float x[SS];
#pragma unroll
        for (int s = 0; s < SS; ++s)
            x[s] = emb[(size_t)sid[s] * DTT + d];
        const float w0 = dwk[d * 3 + 0];
        const float w1 = dwk[d * 3 + 1];
        const float w2 = dwk[d * 3 + 2];
        float mb = 0.f, gs = 0.f;
#pragma unroll
        for (int s = 0; s < SS; ++s) {
            const float xm = (s > 0)      ? x[s - 1] : 0.f;
            const float xp = (s < SS - 1) ? x[s + 1] : 0.f;
            const float h  = w0 * xm + w1 * x[s] + w2 * xp;
            gs += h / (1.f + __expf(-h));   // silu
            mb += x[s];
        }
        gsum [(size_t)bl * DTT + d] = gs;
        mbase[(size_t)bl * DTT + d] = mb;
    }
}

// ---------------------------------------------------------------------------
// WMMA GEMM:  C[M,N] = A[M,K] * W[N,K]^T   (f32 in -> bf16 LDS -> f32 acc)
// Block: 256 threads = 8 waves; tile 64(M) x 64(N); K-step 32.
// Wave w: rows 16*(w&3), cols 32*(w>>2) -> two v_wmma_f32_16x16x32_bf16 / step.
// LDS row stride padded to 40 bf16 (80B) -> conflict-free ds_load_b128.
// Fragment layouts follow ISA 7.12.2 (16-bit A and B both interleaved halves).
// ---------------------------------------------------------------------------
template <typename OutT>
__global__ __launch_bounds__(256)
void gemm_xwT_wmma(const float* __restrict__ A,
                   const float* __restrict__ W,
                   OutT* __restrict__ C,
                   int M, int N, int K) {
    constexpr int TM = 64, TN = 64, TK = 32, LDSP = 40;
    __shared__ __align__(16) __bf16 sA[TM * LDSP];
    __shared__ __align__(16) __bf16 sW[TN * LDSP];

    const int tid  = threadIdx.x;
    const int lane = tid & 31;
    const int wave = tid >> 5;
    const int wm   = wave & 3;           // 16-row group within tile
    const int wn   = wave >> 2;          // 32-col group within tile
    const int m0   = blockIdx.y * TM;
    const int n0   = blockIdx.x * TN;

    // global->LDS staging assignment: 8 f32 per thread
    const int lrow = tid >> 2;           // 0..63
    const int lk   = (tid & 3) * 8;      // 0,8,16,24

    // fragment lane mapping
    const int frow  = lane & 15;
    const int khalf = lane >> 4;

    v8f acc0 = {};
    v8f acc1 = {};

    for (int k0 = 0; k0 < K; k0 += TK) {
        // ---- stage A tile (64x32 f32 -> bf16) ----
        {
            const float4* pa = (const float4*)(A + (size_t)(m0 + lrow) * K + k0 + lk);
            const float4 a0 = pa[0], a1 = pa[1];
            v8bf ba;
            ba[0] = (__bf16)a0.x; ba[1] = (__bf16)a0.y; ba[2] = (__bf16)a0.z; ba[3] = (__bf16)a0.w;
            ba[4] = (__bf16)a1.x; ba[5] = (__bf16)a1.y; ba[6] = (__bf16)a1.z; ba[7] = (__bf16)a1.w;
            *(v8bf*)(&sA[lrow * LDSP + lk]) = ba;

            const float4* pw = (const float4*)(W + (size_t)(n0 + lrow) * K + k0 + lk);
            const float4 w0 = pw[0], w1 = pw[1];
            v8bf bw;
            bw[0] = (__bf16)w0.x; bw[1] = (__bf16)w0.y; bw[2] = (__bf16)w0.z; bw[3] = (__bf16)w0.w;
            bw[4] = (__bf16)w1.x; bw[5] = (__bf16)w1.y; bw[6] = (__bf16)w1.z; bw[7] = (__bf16)w1.w;
            *(v8bf*)(&sW[lrow * LDSP + lk]) = bw;
        }
        if (k0 + TK < K) {   // -> global_prefetch_b8 of next K tile
            __builtin_prefetch(A + (size_t)(m0 + lrow) * K + k0 + TK + lk, 0, 1);
            __builtin_prefetch(W + (size_t)(n0 + lrow) * K + k0 + TK + lk, 0, 1);
        }
        __syncthreads();

        // ---- fragments ----
        BFrag fa, fb0, fb1;
        const int ar = wm * 16 + frow;   // A row: halves hold K {kh*8..+7, kh*8+16..+23}
        fa.h[0] = *(const v8bf*)(&sA[ar * LDSP + khalf * 8]);
        fa.h[1] = *(const v8bf*)(&sA[ar * LDSP + khalf * 8 + 16]);

        const int bc0 = wn * 32 + frow;  // B col: halves hold K {kh*8..+7, kh*8+16..+23}
        fb0.h[0] = *(const v8bf*)(&sW[bc0 * LDSP + khalf * 8]);
        fb0.h[1] = *(const v8bf*)(&sW[bc0 * LDSP + khalf * 8 + 16]);
        const int bc1 = bc0 + 16;
        fb1.h[0] = *(const v8bf*)(&sW[bc1 * LDSP + khalf * 8]);
        fb1.h[1] = *(const v8bf*)(&sW[bc1 * LDSP + khalf * 8 + 16]);

        acc0 = __builtin_amdgcn_wmma_f32_16x16x32_bf16(false, fa.v, false, fb0.v,
                                                       (short)0, acc0, false, false);
        acc1 = __builtin_amdgcn_wmma_f32_16x16x32_bf16(false, fa.v, false, fb1.v,
                                                       (short)0, acc1, false, false);
        __syncthreads();
    }

    // ---- store: VGPR r holds M = khalf*8 + r, N = frow (ISA C/D layout) ----
    const int crow = m0 + wm * 16 + khalf * 8;
    const int ccol = n0 + wn * 32 + frow;
#pragma unroll
    for (int r = 0; r < 8; ++r) {
        OutT* dst = C + (size_t)(crow + r) * N + ccol;
        dst[0]  = (OutT)acc0[r];
        dst[16] = (OutT)acc1[r];
    }
}

// ---------------------------------------------------------------------------
// Kernel 3: m = (mbase + ysum)/S, LayerNorm over DT -> q_txt
// ---------------------------------------------------------------------------
__global__ __launch_bounds__(256)
void ln_kernel(const float* __restrict__ mbase, const float* __restrict__ ysum,
               const float* __restrict__ g, const float* __restrict__ bta,
               float* __restrict__ out) {
    const int row = blockIdx.x;
    const int t   = threadIdx.x;
    __shared__ float red[256];
    const size_t base = (size_t)row * DTT;

    const float a = (mbase[base + t]       + ysum[base + t])       * (1.f / SS);
    const float b = (mbase[base + t + 256] + ysum[base + t + 256]) * (1.f / SS);

    red[t] = a + b;
    __syncthreads();
    for (int s = 128; s > 0; s >>= 1) { if (t < s) red[t] += red[t + s]; __syncthreads(); }
    const float mu = red[0] * (1.f / DTT);
    __syncthreads();

    const float da = a - mu, db = b - mu;
    red[t] = da * da + db * db;
    __syncthreads();
    for (int s = 128; s > 0; s >>= 1) { if (t < s) red[t] += red[t + s]; __syncthreads(); }
    const float inv = rsqrtf(red[0] * (1.f / DTT) + 1e-5f);

    out[base + t]       = da * inv * g[t]       + bta[t];
    out[base + t + 256] = db * inv * g[t + 256] + bta[t + 256];
}

// ---------------------------------------------------------------------------
// Kernel 7: banded attention (band +/-16) + softmax + A row + Fv = A*V
// One block (128 threads / 4 waves) per (b,l).
// ---------------------------------------------------------------------------
__global__ __launch_bounds__(128)
void attn_kernel(const float* __restrict__ Q,
                 const __bf16* __restrict__ Kb,
                 const __bf16* __restrict__ Vb,
                 const float* __restrict__ alphap,
                 float* __restrict__ Aout,
                 float* __restrict__ Fv) {
    const int bl = blockIdx.x;
    const int b  = bl >> 8;          // / L
    const int l  = bl & (LL - 1);
    const int c  = (l * (NN - 1)) / (LL - 1);   // trunc, matches linspace->int32
    int nlo = c - 16; if (nlo < 0) nlo = 0;
    int nhi = c + 16; if (nhi > NN - 1) nhi = NN - 1;
    const int cnt = nhi - nlo + 1;              // <= 33

    __shared__ float sQ[DVV];
    __shared__ float sl[40];
    __shared__ float sp[40];
    __shared__ float sinv;

    const int tid   = threadIdx.x;
    const int wave  = tid >> 5;
    const int lane  = tid & 31;
    const float alpha = alphap[0];
    const float scale = 0.044194173824159216f;  // 512^-0.5
    const float binv  = 1.0f / (16.0f + 1e-6f);

    for (int d = tid; d < DVV; d += 128) sQ[d] = Q[(size_t)bl * DVV + d];
    __syncthreads();

    for (int j = wave; j < cnt; j += 4) {
        const __bf16* kr = Kb + ((size_t)b * NN + nlo + j) * DVV;
        float s = 0.f;
        for (int d = lane; d < DVV; d += 32) s += sQ[d] * (float)kr[d];
#pragma unroll
        for (int off = 16; off > 0; off >>= 1) s += __shfl_xor(s, off);
        if (lane == 0)
            sl[j] = s * scale + alpha * (float)(c - (nlo + j)) * binv;
    }
    __syncthreads();

    if (tid == 0) {
        float mx = -3.0e38f;
        for (int j = 0; j < cnt; ++j) mx = fmaxf(mx, sl[j]);
        float sum = 0.f;
        for (int j = 0; j < cnt; ++j) { const float e = __expf(sl[j] - mx); sp[j] = e; sum += e; }
        sinv = 1.f / sum;
    }
    __syncthreads();
    const float inv = sinv;

    // full A row: zeros outside band, softmax inside
    float* arow = Aout + (size_t)bl * NN;
    for (int n = tid; n < NN; n += 128) {
        float v = 0.f;
        if (n >= nlo && n <= nhi) v = sp[n - nlo] * inv;
        arow[n] = v;
    }

    // Fv = sum_j p_j * V[n_j, :]
    for (int d = tid; d < DVV; d += 128) {
        float acc = 0.f;
        for (int j = 0; j < cnt; ++j)
            acc += sp[j] * inv * (float)Vb[((size_t)b * NN + nlo + j) * DVV + d];
        Fv[(size_t)bl * DVV + d] = acc;
    }
}

// ---------------------------------------------------------------------------
// Launcher
// ---------------------------------------------------------------------------
extern "C" void kernel_launch(void* const* d_in, const int* in_sizes, int n_in,
                              void* d_out, int out_size, void* d_ws, size_t ws_size,
                              hipStream_t stream) {
    const float* vis   = (const float*)d_in[0];
    const int*   ids   = (const int*)  d_in[1];
    const float* emb   = (const float*)d_in[2];
    const float* wq    = (const float*)d_in[3];
    const float* wk    = (const float*)d_in[4];
    const float* wv    = (const float*)d_in[5];
    const float* wv2t  = (const float*)d_in[6];
    const float* dwk   = (const float*)d_in[7];
    const float* pwk   = (const float*)d_in[8];   // [512,512,1] contiguous
    const float* lng   = (const float*)d_in[9];
    const float* lnb   = (const float*)d_in[10];
    const float* alpha = (const float*)d_in[11];

    float* out  = (float*)d_out;
    float* cls  = out;                                        // [8192,1024]
    float* Hout = out + (size_t)BLROWS * VOC;                 // [8192, 512]
    float* Aout = out + (size_t)BLROWS * VOC + (size_t)BLROWS * DTT; // [8192,1024]

    // workspace layout (bytes)
    char* ws = (char*)d_ws;
    const size_t F = (size_t)BLROWS * DTT * sizeof(float);    // 16 MB
    float*  gsum  = (float*)(ws);
    float*  mbase = (float*)(ws + F);
    float*  ysum  = (float*)(ws + 2 * F);
    __bf16* Kbf   = (__bf16*)(ws + 3 * F);
    __bf16* Vbf   = (__bf16*)(ws + 3 * F + (size_t)VISROWS * DVV * sizeof(__bf16));
    float*  qtxt  = gsum;   // gsum dead after pointwise GEMM
    float*  Qf    = mbase;  // mbase dead after LN
    float*  Fv    = ysum;   // ysum dead after LN

    // 1. embed + depthwise conv + silu-sum + base mean
    embed_dwconv_kernel<<<BLROWS, 256, 0, stream>>>(ids, emb, dwk, gsum, mbase);

    // 2. ysum = gsum @ pw^T   (mean-commuted pointwise conv)  [8192,512]x[512,512]
    gemm_xwT_wmma<float><<<dim3(DTT / 64, BLROWS / 64), 256, 0, stream>>>(
        gsum, pwk, ysum, BLROWS, DTT, DTT);

    // 3. layernorm -> q_txt
    ln_kernel<<<BLROWS, 256, 0, stream>>>(mbase, ysum, lng, lnb, qtxt);

    // 4. Q = q_txt @ wq^T
    gemm_xwT_wmma<float><<<dim3(DVV / 64, BLROWS / 64), 256, 0, stream>>>(
        qtxt, wq, Qf, BLROWS, DVV, DTT);

    // 5/6. K,V = vis @ wk^T / wv^T  (bf16 output, fits L2 for banded reuse)
    gemm_xwT_wmma<__bf16><<<dim3(DVV / 64, VISROWS / 64), 256, 0, stream>>>(
        vis, wk, Kbf, VISROWS, DVV, DVV);
    gemm_xwT_wmma<__bf16><<<dim3(DVV / 64, VISROWS / 64), 256, 0, stream>>>(
        vis, wv, Vbf, VISROWS, DVV, DVV);

    // 7. banded attention -> A (output) and Fv
    attn_kernel<<<BLROWS, 128, 0, stream>>>(Qf, Kbf, Vbf, alpha, Aout, Fv);

    // 8. H = Fv @ wv2t^T  -> output region 2
    gemm_xwT_wmma<float><<<dim3(DTT / 64, BLROWS / 64), 256, 0, stream>>>(
        Fv, wv2t, Hout, BLROWS, DTT, DVV);

    // 9. cls = H @ emb^T  -> output region 1
    gemm_xwT_wmma<float><<<dim3(VOC / 64, BLROWS / 64), 256, 0, stream>>>(
        Hout, emb, cls, BLROWS, VOC, DTT);
}
